// GTADExtractor_37649683316907
// MI455X (gfx1250) — matrix operand order repaired and checked by reading, not compiled, not run on hardware
//
#include <hip/hip_runtime.h>
#include <hip/hip_bf16.h>
#include <math.h>

// ---------------- problem sizes ----------------
#define BSZ     2
#define IN_C    256
#define WID     128
#define NGRP    32
#define TT      100
#define P_ANCH  10000
#define NJ      36          // ROI(32) + CTX(4) bins
#define KDIM    3600        // NJ * TT
#define KPAD    3616        // padded to multiple of 32 for WMMA bf16 K=32
#define NCOL    256         // BSZ * OUT_CH
#define OUT_C   128

typedef __attribute__((ext_vector_type(16))) __bf16 v16bf;
typedef __attribute__((ext_vector_type(8)))  __bf16 v8bf;
typedef __attribute__((ext_vector_type(8)))  float  v8f;

__device__ __forceinline__ unsigned short f2bf(float f) {
  union { float f; unsigned u; } v; v.f = f;
  unsigned r = v.u + 0x7FFFu + ((v.u >> 16) & 1u);   // round-to-nearest-even
  return (unsigned short)(r >> 16);
}
__device__ __forceinline__ float relu(float x) { return x > 0.f ? x : 0.f; }

// ---------------- t-branch ----------------
// t1[b,o,t] = relu(t1_b[o] + sum_c t1_w[o,c] * x[b,c,t])
__global__ void k_t1(const float* __restrict__ x, const float* __restrict__ w,
                     const float* __restrict__ bias, float* __restrict__ t1) {
  int idx = blockIdx.x * blockDim.x + threadIdx.x;          // BSZ*WID*TT = 25600
  int b = idx / (WID * TT); int r = idx % (WID * TT);
  int o = r / TT, t = r % TT;
  const float* xb = x + ((size_t)b * IN_C) * TT + t;
  const float* wo = w + (size_t)o * IN_C;
  float acc = bias[o];
  for (int c = 0; c < IN_C; ++c) acc += wo[c] * xb[(size_t)c * TT];
  t1[idx] = relu(acc);
}

// grouped conv1d k=3, pad 1, 32 groups of 4 in / 4 out
__global__ void k_t2(const float* __restrict__ t1, const float* __restrict__ w,
                     const float* __restrict__ bias, float* __restrict__ t2) {
  int idx = blockIdx.x * blockDim.x + threadIdx.x;          // 25600
  int b = idx / (WID * TT); int r = idx % (WID * TT);
  int o = r / TT, t = r % TT;
  int g = o >> 2;
  float acc = bias[o];
  for (int i = 0; i < 4; ++i) {
    const float* in = t1 + ((size_t)b * WID + g * 4 + i) * TT;
    const float* wk = w + ((size_t)o * 4 + i) * 3;
    for (int dk = 0; dk < 3; ++dk) {
      int tt = t + dk - 1;
      if (tt >= 0 && tt < TT) acc += wk[dk] * in[tt];
    }
  }
  t2[idx] = relu(acc);
}

// t3pre[b,o,t] = t3_b[o] + sum_c t3_w[o,c] * t2[b,c,t]   (no relu)
__global__ void k_t3(const float* __restrict__ t2, const float* __restrict__ w,
                     const float* __restrict__ bias, float* __restrict__ t3) {
  int idx = blockIdx.x * blockDim.x + threadIdx.x;          // BSZ*IN_C*TT = 51200
  int b = idx / (IN_C * TT); int r = idx % (IN_C * TT);
  int o = r / TT, t = r % TT;
  const float* in = t2 + ((size_t)b * WID) * TT + t;
  const float* wo = w + (size_t)o * WID;
  float acc = bias[o];
  for (int c = 0; c < WID; ++c) acc += wo[c] * in[(size_t)c * TT];
  t3[idx] = acc;
}

// ---------------- kNN (k=3, self included; d = -||xi-xj||^2, ties -> lowest j) --
__global__ void k_knn(const float* __restrict__ src, int* __restrict__ idx3) {
  int idx = blockIdx.x * blockDim.x + threadIdx.x;          // BSZ*TT = 200
  if (idx >= BSZ * TT) return;
  int b = idx / TT, i = idx % TT;
  const float* X = src + (size_t)b * IN_C * TT;
  float sqi = 0.f;
  for (int c = 0; c < IN_C; ++c) { float v = X[(size_t)c * TT + i]; sqi += v * v; }
  float d0 = -1e30f, d1 = -1e30f, d2 = -1e30f; int i0 = 0, i1 = 0, i2 = 0;
  for (int j = 0; j < TT; ++j) {
    float dot = 0.f, sqj = 0.f;
    for (int c = 0; c < IN_C; ++c) {
      float a = X[(size_t)c * TT + i], bb = X[(size_t)c * TT + j];
      dot += a * bb; sqj += bb * bb;
    }
    float dd = 2.f * dot - sqi - sqj;
    if (dd > d0)      { d2 = d1; i2 = i1; d1 = d0; i1 = i0; d0 = dd; i0 = j; }
    else if (dd > d1) { d2 = d1; i2 = i1; d1 = dd; i1 = j; }
    else if (dd > d2) { d2 = dd; i2 = j; }
  }
  idx3[idx * 3 + 0] = i0; idx3[idx * 3 + 1] = i1; idx3[idx * 3 + 2] = i2;
}

// ---------------- fused s-branch + h ----------------
// one block per (b,n), 128 threads
__global__ void k_s_h(const float* __restrict__ x, const int* __restrict__ idxx,
                      const float* __restrict__ s1w, const float* __restrict__ s1b,
                      const float* __restrict__ s2w, const float* __restrict__ s2b,
                      const float* __restrict__ s3w, const float* __restrict__ s3b,
                      const float* __restrict__ t3pre, float* __restrict__ h) {
  __shared__ float s1buf[3][WID];
  __shared__ float s2buf[3][WID];
  int b = blockIdx.x / TT, n = blockIdx.x % TT;
  int tid = threadIdx.x;
  const float* xb = x + (size_t)b * IN_C * TT;
  const int* nb = idxx + ((size_t)b * TT + n) * 3;
  // s1 = relu(W1 [concat(nb, ctr)] + b1)
  const float* w = s1w + (size_t)tid * (2 * IN_C);
  for (int kk = 0; kk < 3; ++kk) {
    int j = nb[kk];
    float acc = s1b[tid];
    for (int c = 0; c < IN_C; ++c) acc += w[c] * xb[(size_t)c * TT + j];
    for (int c = 0; c < IN_C; ++c) acc += w[IN_C + c] * xb[(size_t)c * TT + n];
    s1buf[kk][tid] = relu(acc);
  }
  __syncthreads();
  // grouped 4x4 linear
  int gb = (tid >> 2) << 2;
  for (int kk = 0; kk < 3; ++kk) {
    float acc = s2b[tid];
    for (int i = 0; i < 4; ++i) acc += s2w[tid * 4 + i] * s1buf[kk][gb + i];
    s2buf[kk][tid] = relu(acc);
  }
  __syncthreads();
  // s3 (128->256), max over k, h = relu(t3 + x + smax)
  for (int half = 0; half < 2; ++half) {
    int c = tid + half * WID;
    float smax = -1e30f;
    for (int kk = 0; kk < 3; ++kk) {
      float acc = s3b[c];
      const float* wc = s3w + (size_t)c * WID;
      for (int m = 0; m < WID; ++m) acc += wc[m] * s2buf[kk][m];
      smax = fmaxf(smax, acc);
    }
    size_t oi = ((size_t)b * IN_C + c) * TT + n;
    h[oi] = relu(t3pre[oi] + x[oi] + smax);
  }
}

// g[b,c,n] = mean of h over 3 NN of h
__global__ void k_g(const float* __restrict__ h, const int* __restrict__ idxh,
                    float* __restrict__ g) {
  int idx = blockIdx.x * blockDim.x + threadIdx.x;          // 51200
  int b = idx / (IN_C * TT); int r = idx % (IN_C * TT);
  int c = r / TT, n = r % TT;
  const int* nb = idxh + ((size_t)b * TT + n) * 3;
  const float* hb = h + ((size_t)b * IN_C + c) * TT;
  g[idx] = (hb[nb[0]] + hb[nb[1]] + hb[nb[2]]) * (1.f / 3.f);
}

// ---------------- M[n][k] (bf16), n = b*128+o, k = j*100+t ----------------
__global__ void k_M(const float* __restrict__ h, const float* __restrict__ g,
                    const float* __restrict__ cw, unsigned short* __restrict__ M) {
  int idx = blockIdx.x * blockDim.x + threadIdx.x;          // NCOL*KPAD = 925696
  if (idx >= NCOL * KPAD) return;
  int n = idx / KPAD, k = idx - n * KPAD;
  float acc = 0.f;
  if (k < KDIM) {
    int j = k / TT, t = k - j * TT;
    int b = n >> 7, o = n & 127;
    const float* H = (j < 32 ? h : g) + ((size_t)b * IN_C) * TT + t;
    const float* cwo = cw + (size_t)o * (IN_C * NJ) + j;
    for (int c = 0; c < IN_C; ++c) acc += cwo[c * NJ] * H[(size_t)c * TT];
  }
  M[idx] = f2bf(acc);
}

// ---------------- W_all[p][k] (bf16) generated from the anchor/ROI-align math --
__global__ void k_wgen(unsigned short* __restrict__ W) {
  __shared__ float row[KPAD];
  int p = blockIdx.x;                        // p = d*100 + s
  int d = p / TT, s = p % TT;
  for (int k = threadIdx.x; k < KPAD; k += blockDim.x) row[k] = 0.f;
  __syncthreads();
  int j = threadIdx.x;
  if (j < NJ) {
    float start = 0.f, endv = 0.f;
    int e = s + d + 1;
    if (e <= TT) { float cl = (float)(d + 1); start = (float)s - cl * 0.5f; endv = (float)e + cl * 0.5f; }
    float width = fmaxf(endv - start, 1.f);
    int out_size = (j < 32) ? 32 : 4;
    int ow = (j < 32) ? j : j - 32;
    float bsz = width / (float)out_size;
    int grid = (int)ceilf(width / (float)out_size); if (grid < 1) grid = 1;
    float invg = 1.f / (float)grid;
    int kb = j * TT;
    for (int i = 0; i < grid; ++i) {
      float xpos = start + (float)ow * bsz + ((float)i + 0.5f) * bsz * invg;
      if (xpos < -1.f || xpos > (float)TT) continue;
      float xc = fmaxf(xpos, 0.f);
      int lo = (int)floorf(xc);
      bool top = lo >= TT - 1;
      if (lo > TT - 1) lo = TT - 1;
      int hi = top ? TT - 1 : lo + 1;
      if (top) xc = (float)lo;
      float lx = xc - (float)lo;
      row[kb + lo] += (1.f - lx) * invg;
      row[kb + hi] += lx * invg;
    }
  }
  __syncthreads();
  unsigned short* out = W + (size_t)p * KPAD;
  for (int k = threadIdx.x; k < KPAD; k += blockDim.x) out[k] = f2bf(row[k]);
}

// ---------------- flagship WMMA GEMM: out[n,p] = relu(cb + W_all(PxK) . M^T(KxN)) --
// One wave per 16x64 tile: 4 accumulators share each A fragment (4 WMMAs per A load,
// 4x less L2 traffic on the 72MB W matrix). 4 waves/block cover all N=256.
// A: lane%16=row, lane-half = K lo/hi 16.
// B: lane%16=col, lane-half = K lo/hi 16. D: VGPR r = row r (+8 for hi lane-half).
__global__ void k_gemm(const unsigned short* __restrict__ W,
                       const unsigned short* __restrict__ M,
                       const float* __restrict__ cb, float* __restrict__ out) {
  int wave = threadIdx.x >> 5;
  int lane = threadIdx.x & 31;
  int p0 = blockIdx.x * 16;
  int n0 = wave * 64;                       // 4 waves -> n in [0,256)
  int rr = lane & 15;
  bool hiH = lane >= 16;
  const unsigned short* Arow = W + (size_t)(p0 + rr) * KPAD + (hiH ? 16 : 0);
  const unsigned short* B0 = M + (size_t)(n0 + rr) * KPAD + (hiH ? 16 : 0);
  const unsigned short* B1 = B0 + (size_t)16 * KPAD;
  const unsigned short* B2 = B0 + (size_t)32 * KPAD;
  const unsigned short* B3 = B0 + (size_t)48 * KPAD;
  v8f acc0 = {}, acc1 = {}, acc2 = {}, acc3 = {};
  for (int kk = 0; kk < KPAD; kk += 32) {
    __builtin_prefetch(Arow + kk + 512, 0, 1);
    v8bf alo = *(const v8bf*)(Arow + kk);
    v8bf ahi = *(const v8bf*)(Arow + kk + 8);
    v16bf a = __builtin_shufflevector(alo, ahi,
                 0, 1, 2, 3, 4, 5, 6, 7, 8, 9, 10, 11, 12, 13, 14, 15);
    v16bf b0 = *(const v16bf*)(B0 + kk);
    v16bf b1 = *(const v16bf*)(B1 + kk);
    v16bf b2 = *(const v16bf*)(B2 + kk);
    v16bf b3 = *(const v16bf*)(B3 + kk);
    acc0 = __builtin_amdgcn_wmma_f32_16x16x32_bf16(false, a, false, b0, (short)0, acc0, false, false);
    acc1 = __builtin_amdgcn_wmma_f32_16x16x32_bf16(false, a, false, b1, (short)0, acc1, false, false);
    acc2 = __builtin_amdgcn_wmma_f32_16x16x32_bf16(false, a, false, b2, (short)0, acc2, false, false);
    acc3 = __builtin_amdgcn_wmma_f32_16x16x32_bf16(false, a, false, b3, (short)0, acc3, false, false);
  }
  int pbase = p0 + (hiH ? 8 : 0);
#pragma unroll
  for (int q = 0; q < 4; ++q) {
    v8f acc = (q == 0) ? acc0 : (q == 1) ? acc1 : (q == 2) ? acc2 : acc3;
    int n = n0 + q * 16 + (lane & 15);
    float bias = cb[n & 127];
    size_t base = (size_t)n * P_ANCH + pbase;
#pragma unroll
    for (int r = 0; r < 8; ++r) out[base + r] = relu(acc[r] + bias);
  }
}

// ---------------- launch ----------------
extern "C" void kernel_launch(void* const* d_in, const int* in_sizes, int n_in,
                              void* d_out, int out_size, void* d_ws, size_t ws_size,
                              hipStream_t stream) {
  const float* x   = (const float*)d_in[0];
  const float* t1w = (const float*)d_in[1];  const float* t1b = (const float*)d_in[2];
  const float* t2w = (const float*)d_in[3];  const float* t2b = (const float*)d_in[4];
  const float* t3w = (const float*)d_in[5];  const float* t3b = (const float*)d_in[6];
  const float* s1w = (const float*)d_in[7];  const float* s1b = (const float*)d_in[8];
  const float* s2w = (const float*)d_in[9];  const float* s2b = (const float*)d_in[10];
  const float* s3w = (const float*)d_in[11]; const float* s3b = (const float*)d_in[12];
  const float* cw  = (const float*)d_in[13]; const float* cb  = (const float*)d_in[14];
  float* out = (float*)d_out;

  char* ws = (char*)d_ws;
  size_t off = 0;
  auto take = [&](size_t bytes) { char* p = ws + off; off = (off + bytes + 255) & ~(size_t)255; return p; };
  float* t1    = (float*)take((size_t)BSZ * WID  * TT * 4);
  float* t2    = (float*)take((size_t)BSZ * WID  * TT * 4);
  float* t3pre = (float*)take((size_t)BSZ * IN_C * TT * 4);
  float* h     = (float*)take((size_t)BSZ * IN_C * TT * 4);
  float* g     = (float*)take((size_t)BSZ * IN_C * TT * 4);
  int*   idxx  = (int*)  take((size_t)BSZ * TT * 3 * 4);
  int*   idxh  = (int*)  take((size_t)BSZ * TT * 3 * 4);
  unsigned short* M = (unsigned short*)take((size_t)NCOL   * KPAD * 2);
  unsigned short* W = (unsigned short*)take((size_t)P_ANCH * KPAD * 2);   // ~72 MB

  k_t1 <<<(BSZ * WID * TT) / 256, 256, 0, stream>>>(x, t1w, t1b, t1);
  k_t2 <<<(BSZ * WID * TT) / 256, 256, 0, stream>>>(t1, t2w, t2b, t2);
  k_t3 <<<(BSZ * IN_C * TT) / 256, 256, 0, stream>>>(t2, t3w, t3b, t3pre);
  k_knn<<<1, 256, 0, stream>>>(x, idxx);
  k_s_h<<<BSZ * TT, WID, 0, stream>>>(x, idxx, s1w, s1b, s2w, s2b, s3w, s3b, t3pre, h);
  k_knn<<<1, 256, 0, stream>>>(h, idxh);
  k_g  <<<(BSZ * IN_C * TT) / 256, 256, 0, stream>>>(h, idxh, g);
  k_M  <<<(NCOL * KPAD + 255) / 256, 256, 0, stream>>>(h, g, cw, M);
  k_wgen<<<P_ANCH, 64, 0, stream>>>(W);
  k_gemm<<<P_ANCH / 16, 128, 0, stream>>>(W, M, cb, out);
  (void)in_sizes; (void)n_in; (void)out_size; (void)ws_size;
}
